// VQTokenizer_80092550136305
// MI455X (gfx1250) — compile-verified, matrix operand order checked by reading, not run on hardware
//
#include <hip/hip_runtime.h>
#include <hip/hip_bf16.h>
#include <math.h>

// Problem constants (from reference)
#define N_TOK  65536
#define D_DIM  128
#define K_CODE 2048

typedef __attribute__((ext_vector_type(16))) _Float16 v16h;
typedef __attribute__((ext_vector_type(8)))  _Float16 v8h;
typedef __attribute__((ext_vector_type(8)))  float    v8f;

// Load one 16x32 f16 WMMA fragment (A or B operand) for wave32.
// Per ISA 7.12.2: lane L (half h = L/16, r = L%16) holds row r,
// K = h*8 + 0..7 in VGPR0-3 and K = 16 + h*8 + 0..7 in VGPR4-7.
// => two contiguous 16-byte chunks per lane.
__device__ __forceinline__ v16h load_frag16x32(const _Float16* __restrict__ p) {
  v16h f;
  const v8h lo = *reinterpret_cast<const v8h*>(p);
  const v8h hi = *reinterpret_cast<const v8h*>(p + 16);
#pragma unroll
  for (int i = 0; i < 8; ++i) { f[i] = lo[i]; f[8 + i] = hi[i]; }
  return f;
}

// ---------------------------------------------------------------------------
// Init: f32 codebook -> f16 copy; zero accumulators + code_count output.
__global__ void vq_init(const float* __restrict__ codebook,
                        _Float16* __restrict__ cbh,
                        float* __restrict__ avgProbSum,
                        float* __restrict__ scal,
                        float* __restrict__ code_count) {
  int i = blockIdx.x * blockDim.x + threadIdx.x;
  if (i < K_CODE * D_DIM) cbh[i] = (_Float16)codebook[i];
  if (i < K_CODE) { avgProbSum[i] = 0.0f; code_count[i] = 0.0f; }
  if (i < 4) scal[i] = 0.0f;   // [0]=msum [1]=eLat [2]=sEnt
}

// ---------------------------------------------------------------------------
// Normalize rows of x -> f16, and accumulate sum(mask). One wave per row.
__global__ void vq_normalize(const float* __restrict__ x,
                             const float* __restrict__ mask,
                             _Float16* __restrict__ xh,
                             float* __restrict__ scal) {
  const int wave = threadIdx.x >> 5;
  const int lane = threadIdx.x & 31;
  const int row  = blockIdx.x * 8 + wave;
  const float* xr = x + (size_t)row * D_DIM;
  const float m   = mask[row];
  const float add = (1.0f - m) * 1e-6f;
  float v[4], ss = 0.0f;
#pragma unroll
  for (int i = 0; i < 4; ++i) { v[i] = xr[lane + 32 * i] + add; ss += v[i] * v[i]; }
#pragma unroll
  for (int off = 16; off > 0; off >>= 1) ss += __shfl_xor(ss, off, 32);
  const float inv = 1.0f / fmaxf(sqrtf(ss), 1e-6f);
  _Float16* xo = xh + (size_t)row * D_DIM;
#pragma unroll
  for (int i = 0; i < 4; ++i) xo[lane + 32 * i] = (_Float16)(v[i] * inv);
  if (lane == 0) atomicAdd(&scal[0], m);
}

// ---------------------------------------------------------------------------
// Pass 1: GEMM (xn @ cbT via WMMA f16->f32) fused with row max + argmax only
// (no transcendentals). Writes rowM (softmax max), indices, histogram,
// e/q latent partials, quantized gather. Each wave: 16 rows x all 2048 codes.
// Per-row affine: a = s*cA + cB,  cA = 200*mask, cB = -200*mask - 1e7*(1-mask)
// (this is (mask*(2s-2) + (1-mask)*(-1e5)) / 0.01, softmax-equivalent).
__global__ void __launch_bounds__(256) vq_pass1(
    const _Float16* __restrict__ xh, const _Float16* __restrict__ cbh,
    const float* __restrict__ mask, const float* __restrict__ codebook,
    float* __restrict__ rowM,
    float* __restrict__ scal, float* __restrict__ code_count,
    float* __restrict__ out_q, int* __restrict__ out_idx) {
  __shared__ int sIdx[128];
  const int wave   = threadIdx.x >> 5;
  const int lane   = threadIdx.x & 31;
  const int half   = lane >> 4;
  const int l16    = lane & 15;
  const int wgRow0 = blockIdx.x * 128;
  const int rBase  = wgRow0 + wave * 16;

  // Preload the 4 A fragments (invariant across all 128 column tiles).
  v16h afr[4];
  const _Float16* arow = xh + (size_t)(rBase + l16) * D_DIM + half * 8;
#pragma unroll
  for (int kk = 0; kk < 4; ++kk) afr[kk] = load_frag16x32(arow + kk * 32);

  float cA[8], cB[8], mrow[8];
#pragma unroll
  for (int j = 0; j < 8; ++j) {
    const float mk = mask[rBase + half * 8 + j];
    mrow[j] = mk;
    cA[j] = 200.0f * mk;
    cB[j] = -200.0f * mk - 1.0e7f * (1.0f - mk);
  }

  float bestA[8];
  int bestIdx[8];
#pragma unroll
  for (int j = 0; j < 8; ++j) { bestA[j] = -__builtin_inff(); bestIdx[j] = 0; }

  for (int t = 0; t < K_CODE / 16; ++t) {
    const int colBase = t * 16;
    const _Float16* brow = cbh + (size_t)(colBase + l16) * D_DIM + half * 8;
    v8f c = {};
#pragma unroll
    for (int kk = 0; kk < 4; ++kk) {
      v16h b = load_frag16x32(brow + kk * 32);
      c = __builtin_amdgcn_wmma_f32_16x16x32_f16(false, afr[kk], false, b,
                                                 (short)0, c, false, false);
    }
    const int col = colBase + l16;
#pragma unroll
    for (int j = 0; j < 8; ++j) {
      const float a = fmaf(c[j], cA[j], cB[j]);
      if (a > bestA[j]) { bestA[j] = a; bestIdx[j] = col; }
    }
  }

  // Reduce across the 16 lanes of each half (same rows, disjoint columns).
#pragma unroll
  for (int off = 1; off < 16; off <<= 1) {
#pragma unroll
    for (int j = 0; j < 8; ++j) {
      const float obA = __shfl_xor(bestA[j], off, 32);
      const int   obI = __shfl_xor(bestIdx[j], off, 32);
      if (obA > bestA[j] || (obA == bestA[j] && obI < bestIdx[j])) {
        bestA[j] = obA; bestIdx[j] = obI;
      }
    }
  }

  if (l16 == 0) {  // lanes 0 and 16 each commit 8 rows
    float eLat = 0.0f;
#pragma unroll
    for (int j = 0; j < 8; ++j) {
      const int r = rBase + half * 8 + j;
      rowM[r] = bestA[j];                    // softmax max == best affinity
      out_idx[r] = bestIdx[j];
      sIdx[wave * 16 + half * 8 + j] = bestIdx[j];
      const float mk = mrow[j];
      const float sBest = 0.005f * bestA[j] + 1.0f;  // invert a = 200 s - 200
      eLat += (2.0f - 2.0f * sBest) * mk;
      atomicAdd(&code_count[bestIdx[j]], mk);
    }
    atomicAdd(&scal[1], eLat);
  }
  __syncthreads();

  // Gather quantized rows (f32 codebook) : 128 rows x 128 floats per WG.
  for (int i = threadIdx.x; i < 128 * 32; i += 256) {
    const int r = i >> 5, c4 = i & 31;
    const float4 v = reinterpret_cast<const float4*>(codebook + (size_t)sIdx[r] * D_DIM)[c4];
    reinterpret_cast<float4*>(out_q + (size_t)(wgRow0 + r) * D_DIM)[c4] = v;
  }
}

// ---------------------------------------------------------------------------
// Pass 2: two GEMM sweeps with the known per-row max M.
// Sweep A: Z = sum exp(a-M), W = sum a*exp(a-M)  -> sample-entropy partials.
// Sweep B: avg_probs column sums with per-row weight mask/Z folded into fma.
__global__ void __launch_bounds__(256) vq_pass2(
    const _Float16* __restrict__ xh, const _Float16* __restrict__ cbh,
    const float* __restrict__ mask,
    const float* __restrict__ rowM,
    float* __restrict__ scal, float* __restrict__ avgProbSum) {
  __shared__ float sProb[K_CODE];
  for (int i = threadIdx.x; i < K_CODE; i += 256) sProb[i] = 0.0f;
  __syncthreads();

  const int wave  = threadIdx.x >> 5;
  const int lane  = threadIdx.x & 31;
  const int half  = lane >> 4;
  const int l16   = lane & 15;
  const int rBase = blockIdx.x * 128 + wave * 16;

  v16h afr[4];
  const _Float16* arow = xh + (size_t)(rBase + l16) * D_DIM + half * 8;
#pragma unroll
  for (int kk = 0; kk < 4; ++kk) afr[kk] = load_frag16x32(arow + kk * 32);

  float cA[8], cB[8], mrow[8], Mr[8];
#pragma unroll
  for (int j = 0; j < 8; ++j) {
    const int r = rBase + half * 8 + j;
    const float mk = mask[r];
    mrow[j] = mk;
    cA[j] = 200.0f * mk;
    cB[j] = -200.0f * mk - 1.0e7f * (1.0f - mk);
    Mr[j] = rowM[r];
  }

  // ---- Sweep A: per-row Z and W ----
  float Zr[8], Wr[8];
#pragma unroll
  for (int j = 0; j < 8; ++j) { Zr[j] = 0.0f; Wr[j] = 0.0f; }

  for (int t = 0; t < K_CODE / 16; ++t) {
    const _Float16* brow = cbh + (size_t)(t * 16 + l16) * D_DIM + half * 8;
    v8f c = {};
#pragma unroll
    for (int kk = 0; kk < 4; ++kk) {
      v16h b = load_frag16x32(brow + kk * 32);
      c = __builtin_amdgcn_wmma_f32_16x16x32_f16(false, afr[kk], false, b,
                                                 (short)0, c, false, false);
    }
#pragma unroll
    for (int j = 0; j < 8; ++j) {
      const float a = fmaf(c[j], cA[j], cB[j]);
      const float e = __expf(a - Mr[j]);
      Zr[j] += e;
      Wr[j] = fmaf(a, e, Wr[j]);
    }
  }
  // Same-scale exps -> plain sum across the 16 lanes of each half.
#pragma unroll
  for (int off = 1; off < 16; off <<= 1) {
#pragma unroll
    for (int j = 0; j < 8; ++j) {
      Zr[j] += __shfl_xor(Zr[j], off, 32);
      Wr[j] += __shfl_xor(Wr[j], off, 32);
    }
  }
  float wgt[8];  // mask / Z  (column-sum weight per row)
#pragma unroll
  for (int j = 0; j < 8; ++j) wgt[j] = mrow[j] / Zr[j];

  if (l16 == 0) {  // sample-entropy partials: sum p log p = W/Z - M - log Z
    float sEnt = 0.0f;
#pragma unroll
    for (int j = 0; j < 8; ++j) {
      const float plogp = Wr[j] / Zr[j] - Mr[j] - __logf(Zr[j]);
      sEnt -= plogp * mrow[j];
    }
    atomicAdd(&scal[2], sEnt);
  }

  // ---- Sweep B: avg_probs column sums (B tiles now hot in WGP$/L2) ----
  for (int t = 0; t < K_CODE / 16; ++t) {
    const int colBase = t * 16;
    const _Float16* brow = cbh + (size_t)(colBase + l16) * D_DIM + half * 8;
    v8f c = {};
#pragma unroll
    for (int kk = 0; kk < 4; ++kk) {
      v16h b = load_frag16x32(brow + kk * 32);
      c = __builtin_amdgcn_wmma_f32_16x16x32_f16(false, afr[kk], false, b,
                                                 (short)0, c, false, false);
    }
    float colSum = 0.0f;
#pragma unroll
    for (int j = 0; j < 8; ++j) {
      const float a = fmaf(c[j], cA[j], cB[j]);
      const float e = __expf(a - Mr[j]);
      colSum = fmaf(e, wgt[j], colSum);
    }
    colSum += __shfl_xor(colSum, 16, 32);   // combine the two row halves
    if (half == 0) atomicAdd(&sProb[colBase + l16], colSum);
  }
  __syncthreads();
  for (int i = threadIdx.x; i < K_CODE; i += 256) atomicAdd(&avgProbSum[i], sProb[i]);
}

// ---------------------------------------------------------------------------
// Finalize the three scalar losses.
__global__ void vq_finalize(const float* __restrict__ scal,
                            const float* __restrict__ avgProbSum,
                            float* __restrict__ out_scalars) {
  __shared__ float red[256];
  const float msum = scal[0];
  float local = 0.0f;
  for (int k = threadIdx.x; k < K_CODE; k += 256) {
    const float ap = avgProbSum[k] / msum;
    local -= ap * __logf(ap + 1e-5f);
  }
  red[threadIdx.x] = local;
  __syncthreads();
  for (int off = 128; off > 0; off >>= 1) {
    if ((int)threadIdx.x < off) red[threadIdx.x] += red[threadIdx.x + off];
    __syncthreads();
  }
  if (threadIdx.x == 0) {
    const float denom = msum + 1e-6f;
    const float avg_entropy = red[0];
    out_scalars[0] = scal[1] / denom;                 // e_latent_loss
    out_scalars[1] = scal[1] / denom;                 // q_latent_loss (identical fwd)
    out_scalars[2] = scal[2] / msum - avg_entropy;    // entropy_loss
  }
}

// ---------------------------------------------------------------------------
extern "C" void kernel_launch(void* const* d_in, const int* in_sizes, int n_in,
                              void* d_out, int out_size, void* d_ws, size_t ws_size,
                              hipStream_t stream) {
  const float* x        = (const float*)d_in[0];
  const float* mask     = (const float*)d_in[1];
  const float* codebook = (const float*)d_in[2];

  // d_out layout (return order): quantized (N*D f32), e_latent, q_latent,
  // entropy (3 f32), code_count (K f32), encoding_indices (N int32 raw).
  float* out         = (float*)d_out;
  float* out_q       = out;
  float* out_scalars = out + (size_t)N_TOK * D_DIM;
  float* out_count   = out_scalars + 3;
  int*   out_idx     = (int*)(out_count + K_CODE);

  // Workspace: ~17.6 MB
  char* ws = (char*)d_ws;
  _Float16* xh  = (_Float16*)ws; ws += (size_t)N_TOK * D_DIM * sizeof(_Float16);
  _Float16* cbh = (_Float16*)ws; ws += (size_t)K_CODE * D_DIM * sizeof(_Float16);
  float* rowM       = (float*)ws; ws += (size_t)N_TOK * sizeof(float);
  float* avgProbSum = (float*)ws; ws += (size_t)K_CODE * sizeof(float);
  float* scal       = (float*)ws;  // 4 floats

  vq_init<<<(K_CODE * D_DIM + 255) / 256, 256, 0, stream>>>(codebook, cbh, avgProbSum,
                                                            scal, out_count);
  vq_normalize<<<N_TOK / 8, 256, 0, stream>>>(x, mask, xh, scal);
  vq_pass1<<<N_TOK / 128, 256, 0, stream>>>(xh, cbh, mask, codebook, rowM,
                                            scal, out_count, out_q, out_idx);
  vq_pass2<<<N_TOK / 128, 256, 0, stream>>>(xh, cbh, mask, rowM, scal, avgProbSum);
  vq_finalize<<<1, 256, 0, stream>>>(scal, avgProbSum, out_scalars);
}
